// TextEncoder_15642270892363
// MI455X (gfx1250) — matrix operand order repaired and checked by reading, not compiled, not run on hardware
//
#include <hip/hip_runtime.h>
#include <hip/hip_bf16.h>
#include <math.h>

// ---------------- problem constants ----------------
constexpr int CN   = 64;     // batch
constexpr int CL   = 510;    // sequence length
constexpr int CD   = 768;    // model dim (K of GEMM)
constexpr int CL1  = 128;    // spans branch 1
constexpr int CK   = 20;     // spans knowledge branch
constexpr int COUT = 300;    // linear out features
constexpr int COUTP= 320;    // padded to 20 tiles of 16 (5 groups of 4 tiles)
constexpr float CEPS = 1e-5f;

constexpr int M1      = CN * CL1;        // 8192
constexpr int MK      = CN * CK;         // 1280
constexpr int MTOT    = M1 + MK;         // 9472  (= 592 * 16, exact tile fit)
constexpr int MTILES  = MTOT / 16;       // 592
constexpr int NGROUPS = COUTP / 64;      // 5  (each wave does a 16x64 strip)

typedef __attribute__((ext_vector_type(16))) _Float16 v16h;
typedef __attribute__((ext_vector_type(8)))  _Float16 v8h;
typedef __attribute__((ext_vector_type(8)))  float    v8f;

union ABfrag { v16h v; v8h h[2]; };

// ---------------- kernel 0: W_lin fp32 -> f16, padded to 320 rows ----------------
__global__ void cvt_w_kernel(const float* __restrict__ Wlin,
                             _Float16* __restrict__ Wh) {
    int i = blockIdx.x * blockDim.x + threadIdx.x;       // over COUTP*CD
    if (i >= COUTP * CD) return;
    int n = i / CD;
    Wh[i] = (n < COUT) ? (_Float16)Wlin[i] : (_Float16)0.0f;
}

// ---------------- kernel 1: ragged span-mean pooling -> f16 (MTOT x 768) -----------
// One block per output row; spans cover at most 4 source rows, so we sum
// directly instead of materializing the reference's (N,L+1,D) cumsum.
// This pass is the HBM-bound part of the problem (~160 MB of f32 activations).
__global__ __launch_bounds__(256)
void pool_kernel(const float* __restrict__ t1,
                 const float* __restrict__ know,
                 const int*   __restrict__ ss,  const int* __restrict__ se,
                 const int*   __restrict__ kss, const int* __restrict__ kse,
                 _Float16*    __restrict__ pooled) {
    int m = blockIdx.x;                                  // 0 .. MTOT-1
    const float* x;
    int s, e;
    if (m < M1) {
        int n = m / CL1;
        x = t1 + (size_t)n * CL * CD;
        s = ss[m];  e = se[m];
    } else {
        int mm = m - M1;
        int n  = mm / CK;
        x = know + (size_t)n * CL * CD;
        s = kss[mm]; e = kse[mm];
    }
    int len = e - s;
    if (len < 1) len = 1;
    float inv = 1.0f / (float)len;
    for (int d = threadIdx.x; d < CD; d += 256) {
        float acc = 0.0f;
        for (int r = s; r < e; ++r) acc += x[(size_t)r * CD + d];
        pooled[(size_t)m * CD + d] = (_Float16)(acc * inv);
    }
}

// ---------------- kernel 2: WMMA GEMM  (MTOT x 768) * (768 x 300) + bias -----------
// One wave32 per 16x64 output strip: the A fragment is loaded once per K-step
// and reused by 4 consecutive v_wmma_f32_16x16x32_f16 into 4 accumulators,
// cutting A-operand L2 traffic ~4x vs one-tile-per-wave.
// Fragment addressing per CDNA5 ISA 7.12.2 VGPR layout tables.
__global__ __launch_bounds__(32)
void wmma_gemm_kernel(const _Float16* __restrict__ Apool,
                      const _Float16* __restrict__ Wh,
                      const float*    __restrict__ blin,
                      float* __restrict__ h1,
                      float* __restrict__ hk) {
    const int m0   = blockIdx.x * 16;
    const int g0   = blockIdx.y * 64;    // first output column of this strip
    const int lane = threadIdx.x;        // 0..31 (wave32)
    const int hi   = lane >> 4;          // lane group 0 / 1
    const int l15  = lane & 15;

    // A: row m0+l15; lane group selects K sub-ranges (+0/+8 and +16/+24)
    const _Float16* Abase = Apool + (size_t)(m0 + l15) * CD;
    // B: column g0+t*16+l15 == that row of Wh; lane group selects K sub-ranges (+0/+8 and +16/+24)
    const _Float16* Bbase = Wh + (size_t)(g0 + l15) * CD;

    v8f acc[4] = {};
#pragma unroll
    for (int k0 = 0; k0 < CD; k0 += 32) {
        ABfrag a;
        a.h[0] = *(const v8h*)(Abase + k0 +      hi * 8);   // K [k0..k0+8)  / [k0+8..16)
        a.h[1] = *(const v8h*)(Abase + k0 + 16 + hi * 8);   // K [k0+16..24) / [k0+24..32)
#pragma unroll
        for (int t = 0; t < 4; ++t) {
            const _Float16* Bb = Bbase + (size_t)t * 16 * CD;
            ABfrag b;
            b.h[0] = *(const v8h*)(Bb + k0 +      hi * 8);  // K [k0..k0+8)  / [k0+8..16)
            b.h[1] = *(const v8h*)(Bb + k0 + 16 + hi * 8);  // K [k0+16..24) / [k0+24..32)
            acc[t] = __builtin_amdgcn_wmma_f32_16x16x32_f16(
                        /*neg_a=*/false, a.v, /*neg_b=*/false, b.v,
                        /*c_mod=*/(short)0, acc[t],
                        /*reuse_a=*/false, /*reuse_b=*/false);
        }
    }

#pragma unroll
    for (int t = 0; t < 4; ++t) {
        const int col = g0 + t * 16 + l15;
        if (col < COUT) {
            const float bias = blin[col];
#pragma unroll
            for (int r = 0; r < 8; ++r) {
                int row = m0 + r + hi * 8;                  // C/D layout: VGPR r -> M = r (+8)
                float val = acc[t][r] + bias;
                if (row < M1) h1[(size_t)row * COUT + col] = val;
                else          hk[(size_t)(row - M1) * COUT + col] = val;
            }
        }
    }
}

// ---------------- kernel 3: in-place LayerNorm(OUT=300) + score dot ---------------
__global__ __launch_bounds__(128)
void ln_score_kernel(float* __restrict__ h1, float* __restrict__ hk,
                     const float* __restrict__ gamma, const float* __restrict__ beta,
                     const float* __restrict__ Wscore, const float* __restrict__ bscore,
                     float* __restrict__ sc) {
    __shared__ float red[128];
    const int m   = blockIdx.x;
    const int tid = threadIdx.x;
    float* row = (m < M1) ? (h1 + (size_t)m * COUT)
                          : (hk + (size_t)(m - M1) * COUT);

    float lsum = 0.0f;
    for (int c = tid; c < COUT; c += 128) lsum += row[c];
    red[tid] = lsum; __syncthreads();
    for (int off = 64; off > 0; off >>= 1) {
        if (tid < off) red[tid] += red[tid + off];
        __syncthreads();
    }
    const float mean = red[0] / (float)COUT;
    __syncthreads();

    float lvar = 0.0f;
    for (int c = tid; c < COUT; c += 128) {
        float d = row[c] - mean; lvar += d * d;
    }
    red[tid] = lvar; __syncthreads();
    for (int off = 64; off > 0; off >>= 1) {
        if (tid < off) red[tid] += red[tid + off];
        __syncthreads();
    }
    const float rstd = rsqrtf(red[0] / (float)COUT + CEPS);
    __syncthreads();

    float lscore = 0.0f;
    for (int c = tid; c < COUT; c += 128) {
        float v = (row[c] - mean) * rstd * gamma[c] + beta[c];
        row[c] = v;
        lscore += v * Wscore[c];
    }
    red[tid] = lscore; __syncthreads();
    for (int off = 64; off > 0; off >>= 1) {
        if (tid < off) red[tid] += red[tid + off];
        __syncthreads();
    }
    if (tid == 0) sc[m] = red[0] + bscore[0];
}

// ---------------- kernel 4: masked softmax over spans (len <= 128) ----------------
__global__ __launch_bounds__(128)
void masked_softmax_kernel(const float* __restrict__ sc,
                           const unsigned char* __restrict__ mask,  // numpy bool (1B)
                           float* __restrict__ out, int len) {
    __shared__ float red[128];
    const int n = blockIdx.x, tid = threadIdx.x;
    float myv = -INFINITY;
    if (tid < len) {
        float s = sc[n * len + tid];
        myv = mask[n * len + tid] ? -INFINITY : s;
    }
    red[tid] = myv; __syncthreads();
    for (int off = 64; off > 0; off >>= 1) {
        if (tid < off) red[tid] = fmaxf(red[tid], red[tid + off]);
        __syncthreads();
    }
    const float mx = red[0]; __syncthreads();
    float ex = (tid < len && myv > -INFINITY) ? expf(myv - mx) : 0.0f;
    red[tid] = ex; __syncthreads();
    for (int off = 64; off > 0; off >>= 1) {
        if (tid < off) red[tid] += red[tid + off];
        __syncthreads();
    }
    if (tid < len) out[n * len + tid] = ex / red[0];
}

// ---------------- kernel 5: broadcast s1 (N,L1) -> score (N,L1,OUT) ---------------
__global__ void broadcast_kernel(const float* __restrict__ s1buf,
                                 float* __restrict__ score) {
    int i = blockIdx.x * blockDim.x + threadIdx.x;
    if (i >= M1 * COUT) return;
    score[i] = s1buf[i / COUT];
}

extern "C" void kernel_launch(void* const* d_in, const int* in_sizes, int n_in,
                              void* d_out, int out_size, void* d_ws, size_t ws_size,
                              hipStream_t stream) {
    // inputs, per setup_inputs() order
    const float* t1        = (const float*)d_in[0];
    const float* know      = (const float*)d_in[1];
    const int*   ss        = (const int*)d_in[2];
    const int*   se        = (const int*)d_in[3];
    const unsigned char* mask1 = (const unsigned char*)d_in[4];   // bool (N,L1)
    const int*   kss       = (const int*)d_in[5];
    const int*   kse       = (const int*)d_in[6];
    const unsigned char* maskk = (const unsigned char*)d_in[7];   // bool (N,K)
    const float* Wlin      = (const float*)d_in[8];
    const float* blin      = (const float*)d_in[9];
    const float* gamma     = (const float*)d_in[10];
    const float* beta      = (const float*)d_in[11];
    const float* Wscore    = (const float*)d_in[12];
    const float* bscore    = (const float*)d_in[13];

    // output regions (return order: h1, score, hk, sk)
    float* h1    = (float*)d_out;
    float* score = h1 + (size_t)M1 * COUT;
    float* hk    = score + (size_t)M1 * COUT;
    float* sk    = hk + (size_t)MK * COUT;

    // workspace layout
    char* ws = (char*)d_ws;
    _Float16* Wh     = (_Float16*)ws;                              // 320*768*2   = 491520 B
    _Float16* pooled = (_Float16*)(ws + 491520);                   // 9472*768*2  = 14548992 B
    float*    sc     = (float*)(ws + 491520 + 14548992);           // 9472*4
    float*    s1buf  = (float*)(ws + 491520 + 14548992 + 37888);   // 8192*4

    // 0) W_lin -> f16 (zero-padded rows 300..319)
    cvt_w_kernel<<<(COUTP * CD + 255) / 256, 256, 0, stream>>>(Wlin, Wh);

    // 1) ragged span-mean pooling -> f16 activations
    pool_kernel<<<MTOT, 256, 0, stream>>>(t1, know, ss, se, kss, kse, pooled);

    // 2) WMMA GEMM (+bias) directly into h1 / hk output regions (16x64 per wave)
    dim3 gg(MTILES, NGROUPS);
    wmma_gemm_kernel<<<gg, 32, 0, stream>>>(pooled, Wh, blin, h1, hk);

    // 3) in-place LayerNorm + score projection
    ln_score_kernel<<<MTOT, 128, 0, stream>>>(h1, hk, gamma, beta, Wscore, bscore, sc);

    // 4) masked softmax: branch1 -> s1buf, knowledge -> sk output
    masked_softmax_kernel<<<CN, 128, 0, stream>>>(sc,      mask1, s1buf, CL1);
    masked_softmax_kernel<<<CN, 128, 0, stream>>>(sc + M1, maskk, sk,    CK);

    // 5) broadcast s1 to (N, L1, OUT) score output
    broadcast_kernel<<<(M1 * COUT + 255) / 256, 256, 0, stream>>>(s1buf, score);
}
